// Flow_14877766714083
// MI455X (gfx1250) — matrix operand order repaired and checked by reading, not compiled, not run on hardware
//
#include <hip/hip_runtime.h>

#define BB 8
#define HH 192
#define WW 192
#define CC 64

typedef __attribute__((ext_vector_type(16))) __bf16 v16bf;
typedef __attribute__((ext_vector_type(8)))  __bf16 v8bf;
typedef __attribute__((ext_vector_type(8)))  float  v8f;

__device__ __forceinline__ float mishf(float x) {
  // x * tanh(softplus(x)); expf overflow -> tanh(inf)=1 -> y=x (correct limit)
  return x * tanhf(log1pf(expf(x)));
}

// Async global->LDS 16-byte transfer (CDNA5 ASYNCcnt path, GVS addressing).
// dst must be a pointer into LDS; low 32 bits of the generic address are the
// LDS byte offset (flat-to-LDS mapping truncates the upper bits).
__device__ __forceinline__ void async_load_b128(const void* base, unsigned goff,
                                                void* ldsdst) {
  unsigned l = (unsigned)(unsigned long long)ldsdst;
  asm volatile("global_load_async_to_lds_b128 %0, %1, %2"
               :: "v"(l), "v"(goff), "s"(base) : "memory");
}

__device__ __forceinline__ void wait_async_all() {
  asm volatile("s_wait_asynccnt 0x0" ::: "memory");
}

// ---------------------------------------------------------------------------
// K1: cost volume (81 shifts, mean over 64 ch, leaky-relu 0.1) + concat
//     [cost | prv | nxt | zero-pad] -> x0 bf16, 224 channels per pixel.
// Tile: 32 (x) by 4 (y) pixels per block of 128 threads (one pixel/thread).
// nxt halo tile streams into LDS via async loads, padded stride 68 f32.
// ---------------------------------------------------------------------------
__global__ void __launch_bounds__(128) cost_concat_kernel(
    const float* __restrict__ prv, const float* __restrict__ nxt,
    __bf16* __restrict__ x0buf)
{
  constexpr int NSTR = 68;                 // padded f32 stride per pixel slot
  __shared__ __align__(16) float s_n[12 * 40 * NSTR];   // 130,560 B
  const int tid = threadIdx.x;
  const int bb = blockIdx.z, by = blockIdx.y, bx = blockIdx.x;
  const int y0 = by * 4, tx0 = bx * 32;

  // cooperative halo load of nxt: rows y0-4..y0+7, cols tx0-4..tx0+35
  for (int t = tid; t < 12 * 40 * 16; t += 128) {
    int c4 = t & 15;
    int rx = (t >> 4) % 40;
    int ry = t / (16 * 40);
    int gy = y0 - 4 + ry, gx = tx0 - 4 + rx;
    float* dst = s_n + (ry * 40 + rx) * NSTR + c4 * 4;
    if (gy >= 0 && gy < HH && gx >= 0 && gx < WW) {
      unsigned goff = (unsigned)((((unsigned)bb * HH + gy) * WW + gx) * CC + c4 * 4) * 4u;
      async_load_b128(nxt, goff, dst);           // global -> LDS, ASYNCcnt
    } else {
      *(float4*)dst = make_float4(0.f, 0.f, 0.f, 0.f);
    }
  }
  wait_async_all();
  __syncthreads();

  const int px = tid & 31, py = tid >> 5;
  const int gy = y0 + py, gx = tx0 + px;

  float p[CC];
  const float* pg = prv + (((size_t)bb * HH + gy) * WW + gx) * CC;
#pragma unroll
  for (int c4 = 0; c4 < 16; ++c4)
    *(float4*)(p + c4 * 4) = *(const float4*)(pg + c4 * 4);

  __bf16* xo = x0buf + (((size_t)bb * HH + gy) * WW + gx) * 224;

#pragma unroll 1
  for (int s = 0; s < 81; ++s) {
    const int i = s / 9, j = s % 9;                 // shift (i-4, j-4)
    const float* np = s_n + ((py + i) * 40 + (px + j)) * NSTR;
    float acc = 0.f;
#pragma unroll
    for (int c = 0; c < CC; c += 4) {
      float4 q = *(const float4*)(np + c);
      acc += p[c] * q.x + p[c + 1] * q.y + p[c + 2] * q.z + p[c + 3] * q.w;
    }
    acc *= (1.0f / 64.0f);
    acc = acc > 0.f ? acc : 0.1f * acc;             // leaky_relu(0.1)
    xo[s] = (__bf16)acc;
  }

  const float* nc = s_n + ((py + 4) * 40 + (px + 4)) * NSTR;  // zero shift
#pragma unroll 1
  for (int c = 0; c < CC; ++c) {
    xo[81 + c]  = (__bf16)p[c];
    xo[145 + c] = (__bf16)nc[c];
  }
#pragma unroll
  for (int c = 209; c < 224; ++c) xo[c] = (__bf16)0.f;        // K-pad
}

// ---------------------------------------------------------------------------
// K2..K5: fused sepconv layer.
//   depthwise 3x3 (f32 accumulate, bf16 in/out via LDS)
//   -> pointwise GEMM with V_WMMA_F32_16X16X32_BF16
//   -> + bias -> mish -> bf16 store.
// Tile: 16 (x) x 4 (y) pixels. Block 256 = 8 waves:
//   wave w handles M-tile row (w&3) and N-half (w>>2).
// CIN = padded (stored) input channels, CINR = real channels, COUT = outputs.
// ---------------------------------------------------------------------------
template <int CIN, int CINR, int COUT>
__global__ void __launch_bounds__(256) sepconv_kernel(
    const __bf16* __restrict__ xin, const float* __restrict__ dww,
    const float* __restrict__ pww, const float* __restrict__ bias,
    __bf16* __restrict__ xout)
{
  constexpr int TX = 16, TY = 4, HX = TX + 2, HY = TY + 2;
  constexpr int XSTR = CIN + 8;            // bf16 stride (bank de-alias, 16B aligned)
  constexpr int KT = CIN / 32, NT = COUT / 16, NT2 = (NT + 1) / 2;

  __shared__ __align__(16) __bf16 s_x[HY * HX * XSTR];  // input tile + halo
  __shared__ __align__(16) __bf16 s_d[TY * TX * XSTR];  // depthwise output
  __shared__ __align__(32) __bf16 s_w[KT * NT * 512];   // B-fragment-swizzled weights
  __shared__ __align__(16) float  s_dw[9 * CIN];        // depthwise weights

  const int tid = threadIdx.x;
  const int bb = blockIdx.z;
  const int y0 = blockIdx.y * TY, tx0 = blockIdx.x * TX;

  // --- stage 1a: async-stream input tile with 1-pixel halo into LDS ---
  for (int t = tid; t < HY * HX * (CIN / 8); t += 256) {
    int c8 = t % (CIN / 8);
    int rem = t / (CIN / 8);
    int hx = rem % HX, hy = rem / HX;
    int gx = tx0 - 1 + hx, gy = y0 - 1 + hy;
    __bf16* dst = s_x + (hy * HX + hx) * XSTR + c8 * 8;
    if (gx >= 0 && gx < WW && gy >= 0 && gy < HH) {
      unsigned goff = (unsigned)((((unsigned)bb * HH + gy) * WW + gx) * CIN + c8 * 8) * 2u;
      async_load_b128(xin, goff, dst);           // global -> LDS, ASYNCcnt
    } else {
      *(float4*)dst = make_float4(0.f, 0.f, 0.f, 0.f);  // SAME padding
    }
  }
  // --- stage 1b: depthwise weights to LDS, zero-padded CINR..CIN ---
  for (int t = tid; t < 9 * CIN; t += 256) {
    int c = t % CIN, k9 = t / CIN;
    s_dw[t] = (c < CINR) ? dww[k9 * CINR + c] : 0.f;
  }
  // --- stage 1c: pointwise weights -> LDS in WMMA B-fragment order ---
  // b[e] = W[kt*32 + (lane/16)*16 + e][nt*16 + lane%16]; rows >= CINR are zero.
  for (int t = tid; t < KT * NT * 32; t += 256) {
    int lane = t & 31;
    int rem = t >> 5;
    int nt = rem % NT, kt = rem / NT;
    int gq = lane >> 4;
#pragma unroll
    for (int e = 0; e < 16; ++e) {
      int k = kt * 32 + gq * 16 + e;
      int n = nt * 16 + (lane & 15);
      float w = (k < CINR) ? pww[k * COUT + n] : 0.f;
      s_w[(size_t)t * 16 + e] = (__bf16)w;
    }
  }
  wait_async_all();        // tile fully resident in LDS (ASYNCcnt == 0)
  __syncthreads();

  // --- stage 2: depthwise 3x3, f32 accumulate, bf16 to s_d ---
  for (int t = tid; t < TY * TX * (CIN / 8); t += 256) {
    int cg = t % (CIN / 8);
    int pp = t / (CIN / 8);
    int px = pp % TX, py = pp / TX;
    float acc[8] = {0.f, 0.f, 0.f, 0.f, 0.f, 0.f, 0.f, 0.f};
#pragma unroll
    for (int dy = 0; dy < 3; ++dy)
#pragma unroll
      for (int dx = 0; dx < 3; ++dx) {
        const __bf16* xp = s_x + ((py + dy) * HX + (px + dx)) * XSTR + cg * 8;
        const float*  wp = s_dw + (dy * 3 + dx) * CIN + cg * 8;
#pragma unroll
        for (int i = 0; i < 8; ++i) acc[i] += (float)xp[i] * wp[i];
      }
    __bf16* dp = s_d + (py * TX + px) * XSTR + cg * 8;
#pragma unroll
    for (int i = 0; i < 8; ++i) dp[i] = (__bf16)acc[i];
  }
  __syncthreads();

  // --- stage 3: pointwise GEMM via WMMA bf16, + bias + mish, store bf16 ---
  const int wv = tid >> 5, lane = tid & 31;
  const int row = wv & 3, nh = wv >> 2;        // M-row and N-half for this wave
  const int g = lane >> 4, nn = lane & 15, m = lane & 15;
  const __bf16* arow = s_d + (row * TX + m) * XSTR;
  const int gy = y0 + row;
  const size_t pixbase = ((size_t)bb * HH + gy) * WW + tx0;

#pragma unroll 1
  for (int t = 0; t < NT2; ++t) {
    const int nt = nh * NT2 + t;
    if (nt >= NT) break;                       // wave-uniform; EXEC stays full
    v8f acc = {0.f, 0.f, 0.f, 0.f, 0.f, 0.f, 0.f, 0.f};
#pragma unroll
    for (int kt = 0; kt < KT; ++kt) {
      const int kb = kt * 32;
      // A (16x32 bf16): lane m, group g reads k in [g*16,g*16+16)
      v8bf lo = *(const v8bf*)(arow + kb + g * 16);
      v8bf hi = *(const v8bf*)(arow + kb + g * 16 + 8);
      v16bf af;
#pragma unroll
      for (int i = 0; i < 8; ++i) { af[i] = lo[i]; af[i + 8] = hi[i]; }
      // B (32x16 bf16): contiguous pre-swizzled 32B per lane
      v16bf bf = *(const v16bf*)(s_w + ((kt * NT + nt) * 32 + lane) * 16);
      acc = __builtin_amdgcn_wmma_f32_16x16x32_bf16(false, af, false, bf,
                                                    (short)0, acc, false, false);
    }
    const int c = nt * 16 + nn;
    const float bia = bias[c];
#pragma unroll
    for (int vv = 0; vv < 8; ++vv) {
      const int px = g * 8 + vv;               // D row = M = pixel x
      float y = acc[vv] + bia;
      xout[(pixbase + px) * COUT + c] = (__bf16)mishf(y);
    }
  }
}

// ---------------------------------------------------------------------------
// K6: BatchNorm (inference) + 3x3 conv 16->2 + scale -> f32 output
// ---------------------------------------------------------------------------
__global__ void __launch_bounds__(256) bn_flow_kernel(
    const __bf16* __restrict__ x4, const float* __restrict__ gam,
    const float* __restrict__ bet, const float* __restrict__ mea,
    const float* __restrict__ var, const float* __restrict__ fw,
    float* __restrict__ out)
{
  const int idx = blockIdx.x * 256 + threadIdx.x;
  if (idx >= BB * HH * WW) return;
  const int gx = idx % WW;
  const int gy = (idx / WW) % HH;
  const int bb = idx / (WW * HH);

  float sc[16], sh[16];
#pragma unroll
  for (int c = 0; c < 16; ++c) {
    float s = gam[c] / sqrtf(var[c] + 1e-3f);
    sc[c] = s;
    sh[c] = bet[c] - mea[c] * s;
  }

  float a0 = 0.f, a1 = 0.f;
#pragma unroll
  for (int dy = -1; dy <= 1; ++dy)
#pragma unroll
    for (int dx = -1; dx <= 1; ++dx) {
      const int yy = gy + dy, xx = gx + dx;
      if (yy < 0 || yy >= HH || xx < 0 || xx >= WW) continue;
      const __bf16* xp = x4 + (((size_t)bb * HH + yy) * WW + xx) * 16;
      const float*  wp = fw + ((dy + 1) * 3 + (dx + 1)) * 32;  // (3,3,16,2)
#pragma unroll
      for (int c = 0; c < 16; ++c) {
        float v = (float)xp[c] * sc[c] + sh[c];
        a0 += v * wp[c * 2];
        a1 += v * wp[c * 2 + 1];
      }
    }
  const float SCALE = 271.52900386f;           // sqrt(192^2 + 192^2)
  out[(size_t)idx * 2]     = SCALE * a0;
  out[(size_t)idx * 2 + 1] = SCALE * a1;
}

// ---------------------------------------------------------------------------
extern "C" void kernel_launch(void* const* d_in, const int* in_sizes, int n_in,
                              void* d_out, int out_size, void* d_ws, size_t ws_size,
                              hipStream_t stream) {
  (void)in_sizes; (void)n_in; (void)out_size; (void)ws_size;
  const float* prv = (const float*)d_in[0];
  const float* nxt = (const float*)d_in[1];
  const float* dw0 = (const float*)d_in[2];
  const float* pw0 = (const float*)d_in[3];
  const float* b0  = (const float*)d_in[4];
  const float* dw1 = (const float*)d_in[5];
  const float* pw1 = (const float*)d_in[6];
  const float* b1  = (const float*)d_in[7];
  const float* dw2 = (const float*)d_in[8];
  const float* pw2 = (const float*)d_in[9];
  const float* b2  = (const float*)d_in[10];
  const float* dw3 = (const float*)d_in[11];
  const float* pw3 = (const float*)d_in[12];
  const float* b3  = (const float*)d_in[13];
  const float* gam = (const float*)d_in[14];
  const float* bet = (const float*)d_in[15];
  const float* mea = (const float*)d_in[16];
  const float* var = (const float*)d_in[17];
  const float* fw  = (const float*)d_in[18];

  char* ws = (char*)d_ws;
  __bf16* bufA = (__bf16*)ws;                                   // x0(224) / x2(64) / x4(16)
  __bf16* bufB = (__bf16*)(ws + (size_t)BB * HH * WW * 224 * 2); // x1(128) / x3(32)

  const dim3 blkCV(128), blk(256);
  cost_concat_kernel<<<dim3(WW / 32, HH / 4, BB), blkCV, 0, stream>>>(prv, nxt, bufA);

  const dim3 grdSC(WW / 16, HH / 4, BB);
  sepconv_kernel<224, 209, 128><<<grdSC, blk, 0, stream>>>(bufA, dw0, pw0, b0, bufB);
  sepconv_kernel<128, 128,  64><<<grdSC, blk, 0, stream>>>(bufB, dw1, pw1, b1, bufA);
  sepconv_kernel< 64,  64,  32><<<grdSC, blk, 0, stream>>>(bufA, dw2, pw2, b2, bufB);
  sepconv_kernel< 32,  32,  16><<<grdSC, blk, 0, stream>>>(bufB, dw3, pw3, b3, bufA);

  bn_flow_kernel<<<dim3((BB * HH * WW + 255) / 256), blk, 0, stream>>>(
      bufA, gam, bet, mea, var, fw, (float*)d_out);
}
